// SACActor_77498389889732
// MI455X (gfx1250) — matrix operand order repaired and checked by reading, not compiled, not run on hardware
//
#include <hip/hip_runtime.h>
#include <hip/hip_bf16.h>
#include <math.h>

#define BB 128
#define SS 512
#define DD 64
#define HH 128
#define GG 512   // 4*H
#define NHEAD 4
#define DHEAD 32
#define NINFO 13
#define NACT 3

typedef __attribute__((ext_vector_type(16))) __bf16 v16bf;
typedef __attribute__((ext_vector_type(8)))  float  v8f;

union FragBF {
  v16bf v;
  float4 f4[2];
};

__device__ __forceinline__ ushort f2bf(float f) {
  unsigned int u = __builtin_bit_cast(unsigned int, f);
  unsigned int r = u + 0x7FFFu + ((u >> 16) & 1u);  // round-to-nearest-even
  return (ushort)(r >> 16);
}
__device__ __forceinline__ float bf2f(ushort h) {
  return __builtin_bit_cast(float, ((unsigned int)h) << 16);
}
__device__ __forceinline__ float fast_rcp(float x) {
  return __builtin_amdgcn_rcpf(x);
}

// ---------------------------------------------------------------------------
// K0a: swizzle [Wx;Wh] (192 x 512) f32 -> bf16 WMMA B-fragment layout.
// B-frag (16-bit, 32x16): lane l (n=l&15, hi=l>>4): halves 0..7 = K[8hi ..),
// halves 8..15 = K[16+8hi ..). Storage [(ntile*6+chunk)*32+lane][16]
// ---------------------------------------------------------------------------
__global__ void prep_weights(const float* __restrict__ Wx,
                             const float* __restrict__ Wh,
                             ushort* __restrict__ wswz) {
  int idx = blockIdx.x * 256 + threadIdx.x;    // 0 .. 98303
  if (idx >= 32 * 6 * 32 * 16) return;
  int kk   = idx & 15;
  int lane = (idx >> 4) & 31;
  int rest = idx >> 9;
  int chunk = rest % 6;
  int ntile = rest / 6;
  int r   = chunk * 32 + ((lane >= 16) ? 8 : 0) + (kk & 7) + ((kk >> 3) << 4);   // K index 0..191
  int col = ntile * 16 + (lane & 15);                    // N index 0..511
  float v = (r < DD) ? Wx[r * GG + col] : Wh[(r - DD) * GG + col];
  wswz[idx] = f2bf(v);
}

// ---------------------------------------------------------------------------
// K0b: pre-swizzle x into A-fragment bf16 layout per (batch-tile, t, chunk).
// A-frag (16-bit, 16x32): lane l (m=l&15, hi=l>>4): halves 0..7 = K[c*32+8hi ..),
// halves 8..15 = K[c*32+16+8hi ..). Storage: [((btile*512+t)*2+chunk)*512 + l*16 + kk]
// => in the LSTM loop each wave loads x-chunks as contiguous 32B per lane.
// ---------------------------------------------------------------------------
__global__ void prep_x(const float* __restrict__ x, ushort* __restrict__ xswz) {
  int idx = blockIdx.x * 256 + threadIdx.x;   // < 8*512*2*512 = 4194304
  if (idx >= 8 * SS * 2 * 512) return;
  int kk   = idx & 15;
  int l    = (idx >> 4) & 31;
  int rest = idx >> 9;
  int chunk = rest & 1;
  int t     = (rest >> 1) & (SS - 1);
  int btile = rest >> 10;
  int hi = l >> 4, m = l & 15;
  int K = chunk * 32 + hi * 8 + (kk & 7) + ((kk >> 3) << 4);
  float v = x[((size_t)(btile * 16 + m) * SS + t) * DD + K];
  xswz[idx] = f2bf(v);
}

// ---------------------------------------------------------------------------
// K1: sLSTM recurrence. grid=8 (16 batch rows each), block=256 (8 waves).
// All 24 B-fragments (192 VGPRs) resident for the whole kernel; x A-frags
// streamed coalesced from global; only h round-trips through (padded) LDS.
// Wave w owns column 16w..16w+15 of all four gates => c/n/h stay in VGPRs.
// ---------------------------------------------------------------------------
#define HPAD (HH + 8)   // 272B row stride: 16B-aligned, 4-bank row skew
__global__ __launch_bounds__(256) void lstm_kernel(
    const ushort* __restrict__ xswz, const float* __restrict__ b_cell,
    const ushort* __restrict__ wswz,
    ushort* __restrict__ seqh, float* __restrict__ hlast) {
  __shared__ __align__(16) ushort hsh[16][HPAD];   // h tile, bf16
  const int tid = threadIdx.x;
  const int w   = tid >> 5;          // wave 0..7
  const int l   = tid & 31;          // lane
  const int hi  = (l >> 4) & 1;
  const int ln  = l & 15;
  const int b0  = blockIdx.x * 16;
  const int col = w * 16 + ln;       // column within each gate (0..127)

  for (int i = tid; i < 16 * HH; i += 256) hsh[i >> 7][i & 127] = 0;

  // B fragments: resident in VGPRs across all 512 steps
  FragBF bfr[4][6];
#pragma unroll
  for (int g = 0; g < 4; ++g)
#pragma unroll
    for (int ch = 0; ch < 6; ++ch) {
      const ushort* bp = wswz + ((size_t)((g * 8 + w) * 6 + ch) * 32 + l) * 16;
      bfr[g][ch].f4[0] = *(const float4*)(bp);
      bfr[g][ch].f4[1] = *(const float4*)(bp + 8);
    }

  const float bi  = b_cell[col];
  const float bfg = b_cell[HH + col];
  const float bz  = b_cell[2 * HH + col];
  const float bo  = b_cell[3 * HH + col];

  v8f c8 = {};
  v8f n8;
#pragma unroll
  for (int e = 0; e < 8; ++e) n8[e] = 1.0f;

  const ushort* xp = xswz + (size_t)blockIdx.x * SS * 1024 + l * 16;
  __syncthreads();   // initial h zeros visible

  for (int t = 0; t < SS; ++t) {
    // ---- load all 6 A fragments (2 from global x-stream, 4 from LDS h) ----
    FragBF a[6];
    const ushort* xt = xp + (size_t)t * 1024;
    a[0].f4[0] = *(const float4*)(xt);
    a[0].f4[1] = *(const float4*)(xt + 8);
    a[1].f4[0] = *(const float4*)(xt + 512);
    a[1].f4[1] = *(const float4*)(xt + 520);
    if (t + 1 < SS) __builtin_prefetch(xt + 1024, 0, 3);
#pragma unroll
    for (int ch = 0; ch < 4; ++ch) {
      const ushort* ar = &hsh[ln][ch * 32 + hi * 8];
      a[2 + ch].f4[0] = *(const float4*)(ar);
      a[2 + ch].f4[1] = *(const float4*)(ar + 16);
    }

    // ---- 24 WMMAs: 4 independent accumulator chains of depth 6 ----
    v8f acc[4] = {{}, {}, {}, {}};
#pragma unroll
    for (int ch = 0; ch < 6; ++ch)
#pragma unroll
      for (int g = 0; g < 4; ++g)
        acc[g] = __builtin_amdgcn_wmma_f32_16x16x32_bf16(
            false, a[ch].v, false, bfr[g][ch].v, (short)0, acc[g], false, false);
    __syncthreads();   // all LDS h reads complete

    // ---- gates + state update (branch-free fast math) ----
#pragma unroll
    for (int e = 0; e < 8; ++e) {
      int m = e + hi * 8;
      float gi = acc[0][e] + bi;
      float gf = acc[1][e] + bfg;
      float gz = acc[2][e] + bz;
      float go = acc[3][e] + bo;
      float iv = __expf(gi);
      float fv = fast_rcp(1.0f + __expf(-gf));
      float zv = 1.0f - 2.0f * fast_rcp(__expf(2.0f * gz) + 1.0f);
      float ov = fast_rcp(1.0f + __expf(-go));
      float cv = fv * c8[e] + iv * zv;
      float nv = fv * n8[e] + iv;
      c8[e] = cv;
      n8[e] = nv;
      float hv = ov * cv * fast_rcp(nv);
      ushort hb = f2bf(hv);
      hsh[m][col] = hb;                                    // next-step A
      seqh[((size_t)(b0 + m) * SS + t) * HH + col] = hb;   // for attention
      if (t == SS - 1) hlast[(size_t)(b0 + m) * HH + col] = hv;
    }
    __syncthreads();   // h writes visible before next step's reads
  }
}

// ---------------------------------------------------------------------------
// K2: last-query attention, fully folded.  One WG per batch element.
//   q = h_last@Wq + bq ;  wt_h = Wk[:,h]@q_h ;  score = h_s.wt_h + q_h.bk_h
//   softmax over s ;  hbar_h = sum att*h_s ;  out = (hbar@Wv + bv)@Wo + bo
// ---------------------------------------------------------------------------
__global__ __launch_bounds__(256) void attn_kernel(
    const ushort* __restrict__ seqh, const float* __restrict__ hlast,
    const float* __restrict__ Wq, const float* __restrict__ bq,
    const float* __restrict__ Wk, const float* __restrict__ bk,
    const float* __restrict__ Wv, const float* __restrict__ bv,
    const float* __restrict__ Wo, const float* __restrict__ bo,
    float* __restrict__ ctx) {
  __shared__ float hl[HH], q[HH], wt[NHEAD][HH], cb[NHEAD];
  __shared__ float sc[NHEAD][SS], hbar[NHEAD][HH], of[HH], red[256];
  const int tid = threadIdx.x;
  const int b = blockIdx.x;

  if (tid < HH) hl[tid] = hlast[(size_t)b * HH + tid];
  __syncthreads();
  if (tid < HH) {
    float s = bq[tid];
    for (int k = 0; k < HH; ++k) s += hl[k] * Wq[k * HH + tid];
    q[tid] = s;
  }
  __syncthreads();
  for (int i = tid; i < NHEAD * HH; i += 256) {
    int h = i >> 7, j = i & 127;
    float s = 0.f;
    for (int d = 0; d < DHEAD; ++d) s += Wk[j * HH + h * DHEAD + d] * q[h * DHEAD + d];
    wt[h][j] = s;
  }
  if (tid < NHEAD) {
    float s = 0.f;
    for (int d = 0; d < DHEAD; ++d) s += bk[tid * DHEAD + d] * q[tid * DHEAD + d];
    cb[tid] = s;
  }
  __syncthreads();

  const float scale = 0.1767766952966369f;  // 1/sqrt(32)
  for (int s0 = tid; s0 < SS; s0 += 256) {
    const ushort* row = seqh + ((size_t)b * SS + s0) * HH;
    float d0 = 0.f, d1 = 0.f, d2 = 0.f, d3 = 0.f;
    for (int j = 0; j < HH; ++j) {
      float v = bf2f(row[j]);
      d0 += v * wt[0][j]; d1 += v * wt[1][j];
      d2 += v * wt[2][j]; d3 += v * wt[3][j];
    }
    sc[0][s0] = (d0 + cb[0]) * scale;
    sc[1][s0] = (d1 + cb[1]) * scale;
    sc[2][s0] = (d2 + cb[2]) * scale;
    sc[3][s0] = (d3 + cb[3]) * scale;
  }
  __syncthreads();

  for (int h = 0; h < NHEAD; ++h) {
    float m = -1e30f;
    for (int s0 = tid; s0 < SS; s0 += 256) m = fmaxf(m, sc[h][s0]);
    red[tid] = m; __syncthreads();
    for (int off = 128; off > 0; off >>= 1) {
      if (tid < off) red[tid] = fmaxf(red[tid], red[tid + off]);
      __syncthreads();
    }
    m = red[0]; __syncthreads();
    float sum = 0.f;
    for (int s0 = tid; s0 < SS; s0 += 256) {
      float e = __expf(sc[h][s0] - m);
      sc[h][s0] = e;
      sum += e;
    }
    red[tid] = sum; __syncthreads();
    for (int off = 128; off > 0; off >>= 1) {
      if (tid < off) red[tid] += red[tid + off];
      __syncthreads();
    }
    float inv = 1.0f / red[0]; __syncthreads();
    for (int s0 = tid; s0 < SS; s0 += 256) sc[h][s0] *= inv;
    __syncthreads();
  }

  for (int it = 0; it < 2; ++it) {
    int h = (tid >> 7) + it * 2;
    int j = tid & 127;
    float s = 0.f;
    for (int s0 = 0; s0 < SS; ++s0)
      s += sc[h][s0] * bf2f(seqh[((size_t)b * SS + s0) * HH + j]);
    hbar[h][j] = s;
  }
  __syncthreads();
  if (tid < HH) {
    int h = tid >> 5;
    float s = bv[tid];
    for (int j = 0; j < HH; ++j) s += hbar[h][j] * Wv[j * HH + tid];
    of[tid] = s;
  }
  __syncthreads();
  if (tid < HH) {
    float s = bo[tid];
    for (int c2 = 0; c2 < HH; ++c2) s += of[c2] * Wo[c2 * HH + tid];
    ctx[(size_t)b * HH + tid] = s;
  }
}

// ---------------------------------------------------------------------------
// K3: head MLP (LayerNorm + exact GELU) -> mu, clipped log_std
// ---------------------------------------------------------------------------
__global__ __launch_bounds__(128) void head_kernel(
    const float* __restrict__ ctx, const float* __restrict__ info,
    const float* __restrict__ W1, const float* __restrict__ b1,
    const float* __restrict__ ln_g, const float* __restrict__ ln_b,
    const float* __restrict__ W2, const float* __restrict__ b2,
    const float* __restrict__ Wmu, const float* __restrict__ bmu,
    const float* __restrict__ Wls, const float* __restrict__ bls,
    float* __restrict__ out) {
  __shared__ float cat[HH + NINFO], a1[128], a2[128], red[128];
  const int tid = threadIdx.x;
  const int b = blockIdx.x;
  if (tid < HH) cat[tid] = ctx[(size_t)b * HH + tid];
  if (tid < NINFO) cat[HH + tid] = info[b * NINFO + tid];
  __syncthreads();

  float s = b1[tid];
  for (int k = 0; k < HH + NINFO; ++k) s += cat[k] * W1[k * 128 + tid];

  red[tid] = s; __syncthreads();
  for (int off = 64; off > 0; off >>= 1) {
    if (tid < off) red[tid] += red[tid + off];
    __syncthreads();
  }
  float mu_ = red[0] * (1.0f / 128.0f); __syncthreads();
  float dv = s - mu_;
  red[tid] = dv * dv; __syncthreads();
  for (int off = 64; off > 0; off >>= 1) {
    if (tid < off) red[tid] += red[tid + off];
    __syncthreads();
  }
  float var = red[0] * (1.0f / 128.0f); __syncthreads();

  float tn = dv * rsqrtf(var + 1e-5f) * ln_g[tid] + ln_b[tid];
  float g1 = 0.5f * tn * (1.0f + erff(tn * 0.70710678118654752f));
  a1[tid] = g1; __syncthreads();

  float s2 = b2[tid];
  for (int k = 0; k < 128; ++k) s2 += a1[k] * W2[k * 128 + tid];
  float g2 = 0.5f * s2 * (1.0f + erff(s2 * 0.70710678118654752f));
  a2[tid] = g2; __syncthreads();

  if (tid < NACT) {
    float sm = bmu[tid], sl = bls[tid];
    for (int k = 0; k < 128; ++k) {
      sm += a2[k] * Wmu[k * NACT + tid];
      sl += a2[k] * Wls[k * NACT + tid];
    }
    out[b * NACT + tid] = sm;
    out[BB * NACT + b * NACT + tid] = fminf(fmaxf(sl, -20.0f), 2.0f);
  }
}

// ---------------------------------------------------------------------------
extern "C" void kernel_launch(void* const* d_in, const int* in_sizes, int n_in,
                              void* d_out, int out_size, void* d_ws, size_t ws_size,
                              hipStream_t stream) {
  const float* x      = (const float*)d_in[0];
  const float* info   = (const float*)d_in[1];
  const float* Wx     = (const float*)d_in[2];
  const float* Wh     = (const float*)d_in[3];
  const float* b_cell = (const float*)d_in[4];
  const float* Wq = (const float*)d_in[5];  const float* bq = (const float*)d_in[6];
  const float* Wk = (const float*)d_in[7];  const float* bk = (const float*)d_in[8];
  const float* Wv = (const float*)d_in[9];  const float* bv = (const float*)d_in[10];
  const float* Wo = (const float*)d_in[11]; const float* bo = (const float*)d_in[12];
  const float* W1 = (const float*)d_in[13]; const float* b1 = (const float*)d_in[14];
  const float* lng = (const float*)d_in[15]; const float* lnb = (const float*)d_in[16];
  const float* W2 = (const float*)d_in[17]; const float* b2 = (const float*)d_in[18];
  const float* Wmu = (const float*)d_in[19]; const float* bmu = (const float*)d_in[20];
  const float* Wls = (const float*)d_in[21]; const float* bls = (const float*)d_in[22];

  char* ws = (char*)d_ws;
  ushort* wswz  = (ushort*)(ws);                       // 196608 B
  ushort* xswz  = (ushort*)(ws + 196608);              // 8388608 B
  ushort* seqh  = (ushort*)(ws + 8585216);             // 16777216 B
  float*  hlast = (float*)(ws + 25362432);             // 65536 B
  float*  ctx   = (float*)(ws + 25427968);             // 65536 B
  float*  out = (float*)d_out;

  prep_weights<<<384, 256, 0, stream>>>(Wx, Wh, wswz);
  prep_x<<<16384, 256, 0, stream>>>(x, xswz);
  lstm_kernel<<<8, 256, 0, stream>>>(xswz, b_cell, wswz, seqh, hlast);
  attn_kernel<<<128, 256, 0, stream>>>(seqh, hlast, Wq, bq, Wk, bk, Wv, bv,
                                       Wo, bo, ctx);
  head_kernel<<<128, 128, 0, stream>>>(ctx, info, W1, b1, lng, lnb, W2, b2,
                                       Wmu, bmu, Wls, bls, out);
}
